// EGNNLayer_68461778698587
// MI455X (gfx1250) — matrix operand order repaired and checked by reading, not compiled, not run on hardware
//
#include <hip/hip_runtime.h>
#include <hip/hip_bf16.h>

typedef __bf16 bf16_t;
typedef __attribute__((ext_vector_type(16))) __bf16 v16bf;
typedef __attribute__((ext_vector_type(8)))  __bf16 v8bf;
typedef __attribute__((ext_vector_type(8)))  float  v8f;

#define HID 128

// ---------------------------------------------------------------------------
// A-fragment load (16x32 bf16, MxK) from an LDS row pointer.
// Lanes 0-15 hold M=0..15 (half=0 -> K 0-15), lanes 16-31 hold M=0..15
// (half=1 -> K 16-31). Elements 0..15 are contiguous K (kbase+half*16 ..),
// matching the B-fragment packing of wconv_kernel.
// ---------------------------------------------------------------------------
__device__ inline v16bf load_a_frag(const bf16_t* rowp, int kbase, int half) {
  v8bf lo = *(const v8bf*)(rowp + kbase + half * 16);
  v8bf hi = *(const v8bf*)(rowp + kbase + half * 16 + 8);
  v16bf a;
#pragma unroll
  for (int i = 0; i < 8; ++i) { a[i] = lo[i]; a[8 + i] = hi[i]; }
  return a;
}

__device__ inline float silu_f(float x) { return x / (1.0f + __expf(-x)); }

// ---------------------------------------------------------------------------
// Zero fill
// ---------------------------------------------------------------------------
__global__ void zero_kernel(float* __restrict__ p, long long n) {
  long long i = (long long)blockIdx.x * blockDim.x + threadIdx.x;
  if (i < n) p[i] = 0.0f;
}

// ---------------------------------------------------------------------------
// Repack a KxN(=128) f32 weight matrix into per-lane B-fragment order (bf16).
// B-fragment as packed here (32x16 bf16, KxN, contiguous K): lane half h, col n=lane&15, VGPR j holds
// K = h*16 + 2j, 2j+1. Output index: ((nt*ksteps + ks)*32 + lane)*16 + elem.
// One thread per (nt, ks, lane) -> writes 16 contiguous bf16.
// ---------------------------------------------------------------------------
__global__ void wconv_kernel(const float* __restrict__ W, bf16_t* __restrict__ out,
                             int ksteps) {
  int tid = blockIdx.x * blockDim.x + threadIdx.x;
  int total = 8 * ksteps * 32;
  if (tid >= total) return;
  int lane = tid & 31;
  int ks = (tid >> 5) % ksteps;
  int nt = tid / (32 * ksteps);
  int h_ = lane >> 4;
  int n = lane & 15;
  bf16_t* o = out + (long long)tid * 16;
#pragma unroll
  for (int j = 0; j < 8; ++j) {
    int k = ks * 32 + h_ * 16 + 2 * j;
    o[2 * j]     = (bf16_t)W[(long long)k * HID + nt * 16 + n];
    o[2 * j + 1] = (bf16_t)W[(long long)(k + 1) * HID + nt * 16 + n];
  }
}

// ---------------------------------------------------------------------------
// Edge kernel: 64 edges per block, 4 waves; wave w owns edge rows [16w,16w+16).
//   msg = SiLU([h_src|h_dst|e] @ W1 + b1) @ W2 + b2
//   e_new = e + msg ; gate = sigmoid(e_new . Wg + bg)
//   atomic scatter of msg*gate into agg[dst], degree count into deg[dst]
// ---------------------------------------------------------------------------
__global__ __launch_bounds__(128) void edge_kernel(
    const float* __restrict__ h, const float* __restrict__ e,
    const int* __restrict__ src, const int* __restrict__ dst,
    const bf16_t* __restrict__ W1f, const float* __restrict__ b1,
    const bf16_t* __restrict__ W2f, const float* __restrict__ b2,
    const float* __restrict__ Wg, const float* __restrict__ bg,
    float* __restrict__ out_e, float* __restrict__ agg, float* __restrict__ deg,
    int E) {
  __shared__ __align__(16) bf16_t smA[64 * 200];  // 192-col chunk (+8 pad)
  __shared__ __align__(16) bf16_t smH[64 * 136];  // hidden (128 + 8 pad)

  const int tid = threadIdx.x;
  const int lane = tid & 31;
  const int wave = tid >> 5;   // == M tile
  const int half = lane >> 4;
  const int n = lane & 15;
  const int ebase = blockIdx.x * 64;

  if (tid < 64) {
    int ei = ebase + tid;
    if (ei < E) atomicAdd(deg + dst[ei], 1.0f);
  }

  // Accumulators for the 16x128 output strip of this wave, init with b1.
  v8f acc[8];
#pragma unroll
  for (int nt = 0; nt < 8; ++nt) {
    float bv = b1[nt * 16 + n];
#pragma unroll
    for (int j = 0; j < 8; ++j) acc[nt][j] = bv;
  }

  const int row = wave * 16 + n;  // A-fragment row for this lane

  // GEMM1: K = 384 in two staged chunks of 192 columns.
  for (int ch = 0; ch < 2; ++ch) {
    __syncthreads();
    const int coff = ch * 192;
    for (int idx = tid; idx < 64 * 48; idx += 128) {
      int r = idx / 48, c4 = idx % 48;
      int ei = min(ebase + r, E - 1);
      int c = coff + c4 * 4;
      const float* sp;
      if (c < 128)      sp = h + (long long)src[ei] * HID + c;
      else if (c < 256) sp = h + (long long)dst[ei] * HID + (c - 128);
      else              sp = e + (long long)ei * HID + (c - 256);
      float4 v = *(const float4*)sp;
      bf16_t* dp = smA + r * 200 + c4 * 4;
      dp[0] = (bf16_t)v.x; dp[1] = (bf16_t)v.y;
      dp[2] = (bf16_t)v.z; dp[3] = (bf16_t)v.w;
    }
    __syncthreads();
#pragma unroll
    for (int ks = 0; ks < 6; ++ks) {
      v16bf a = load_a_frag(smA + row * 200, ks * 32, half);
      const int gks = ch * 6 + ks;
#pragma unroll
      for (int nt = 0; nt < 8; ++nt) {
        v16bf b = *(const v16bf*)(W1f + (((long long)nt * 12 + gks) * 32 + lane) * 16);
        acc[nt] = __builtin_amdgcn_wmma_f32_16x16x32_bf16(
            false, a, false, b, (short)0, acc[nt], false, false);
      }
    }
  }

  // SiLU -> bf16 hidden in LDS (D layout: element j -> row j + 8*half).
#pragma unroll
  for (int nt = 0; nt < 8; ++nt) {
#pragma unroll
    for (int j = 0; j < 8; ++j) {
      float s = silu_f(acc[nt][j]);
      smH[(wave * 16 + half * 8 + j) * 136 + nt * 16 + n] = (bf16_t)s;
    }
  }
  __syncthreads();

  // GEMM2: msg = hidden @ W2 + b2 (K = 128).
#pragma unroll
  for (int nt = 0; nt < 8; ++nt) {
    float bv = b2[nt * 16 + n];
    v8f c;
#pragma unroll
    for (int j = 0; j < 8; ++j) c[j] = bv;
#pragma unroll
    for (int ks = 0; ks < 4; ++ks) {
      v16bf a = load_a_frag(smH + row * 136, ks * 32, half);
      v16bf b = *(const v16bf*)(W2f + (((long long)nt * 4 + ks) * 32 + lane) * 16);
      c = __builtin_amdgcn_wmma_f32_16x16x32_bf16(
          false, a, false, b, (short)0, c, false, false);
    }
    acc[nt] = c;  // msg
  }

  // e_new = e + msg, write out, accumulate gate partials (e_new . Wg).
  float part[8];
#pragma unroll
  for (int j = 0; j < 8; ++j) part[j] = 0.0f;
#pragma unroll
  for (int nt = 0; nt < 8; ++nt) {
    float wg = Wg[nt * 16 + n];
#pragma unroll
    for (int j = 0; j < 8; ++j) {
      int er = ebase + wave * 16 + half * 8 + j;
      int erc = min(er, E - 1);
      long long idx = (long long)erc * HID + nt * 16 + n;
      float enew = e[idx] + acc[nt][j];
      if (er < E) out_e[idx] = enew;
      part[j] += enew * wg;
    }
  }
  const float bg0 = bg[0];
  float gate[8];
#pragma unroll
  for (int j = 0; j < 8; ++j) {
    float s = part[j];
    s += __shfl_xor(s, 1);
    s += __shfl_xor(s, 2);
    s += __shfl_xor(s, 4);
    s += __shfl_xor(s, 8);  // reduce across the 16-lane half (wave32)
    gate[j] = 1.0f / (1.0f + __expf(-(s + bg0)));
  }

  // Scatter m = msg * gate into agg[dst].
#pragma unroll
  for (int j = 0; j < 8; ++j) {
    int er = ebase + wave * 16 + half * 8 + j;
    if (er < E) {
      float g = gate[j];
      float* ar = agg + (long long)dst[er] * HID;
#pragma unroll
      for (int nt = 0; nt < 8; ++nt)
        atomicAdd(ar + nt * 16 + n, acc[nt][j] * g);
    }
  }
}

// ---------------------------------------------------------------------------
// Node kernel: 64 nodes per block, 4 waves.
//   h_new = SiLU([h | agg/(deg+1e-6)] @ U1 + c1) @ U2 + c2
// ---------------------------------------------------------------------------
__global__ __launch_bounds__(128) void node_kernel(
    const float* __restrict__ h, const float* __restrict__ agg,
    const float* __restrict__ deg,
    const bf16_t* __restrict__ U1f, const float* __restrict__ c1,
    const bf16_t* __restrict__ U2f, const float* __restrict__ c2,
    float* __restrict__ out_h, int N) {
  __shared__ __align__(16) bf16_t smA[64 * 264];  // 256 cols + 8 pad
  __shared__ __align__(16) bf16_t smH[64 * 136];
  __shared__ float sscale[64];

  const int tid = threadIdx.x;
  const int lane = tid & 31;
  const int wave = tid >> 5;
  const int half = lane >> 4;
  const int n = lane & 15;
  const int nbase = blockIdx.x * 64;

  if (tid < 64) {
    int r = min(nbase + tid, N - 1);
    sscale[tid] = 1.0f / (deg[r] + 1e-6f);
  }
  __syncthreads();

  for (int idx = tid; idx < 64 * 64; idx += 128) {
    int r = idx / 64, c4 = idx % 64;
    int nr = min(nbase + r, N - 1);
    int c = c4 * 4;
    float4 v;
    float scale = 1.0f;
    if (c < 128) {
      v = *(const float4*)(h + (long long)nr * HID + c);
    } else {
      v = *(const float4*)(agg + (long long)nr * HID + (c - 128));
      scale = sscale[r];
    }
    bf16_t* dp = smA + r * 264 + c;
    dp[0] = (bf16_t)(v.x * scale); dp[1] = (bf16_t)(v.y * scale);
    dp[2] = (bf16_t)(v.z * scale); dp[3] = (bf16_t)(v.w * scale);
  }
  __syncthreads();

  const int row = wave * 16 + n;

  // GEMM1: K = 256 with U1, then SiLU -> smH.
#pragma unroll
  for (int nt = 0; nt < 8; ++nt) {
    float bv = c1[nt * 16 + n];
    v8f c;
#pragma unroll
    for (int j = 0; j < 8; ++j) c[j] = bv;
#pragma unroll
    for (int ks = 0; ks < 8; ++ks) {
      v16bf a = load_a_frag(smA + row * 264, ks * 32, half);
      v16bf b = *(const v16bf*)(U1f + (((long long)nt * 8 + ks) * 32 + lane) * 16);
      c = __builtin_amdgcn_wmma_f32_16x16x32_bf16(
          false, a, false, b, (short)0, c, false, false);
    }
#pragma unroll
    for (int j = 0; j < 8; ++j)
      smH[(wave * 16 + half * 8 + j) * 136 + nt * 16 + n] = (bf16_t)silu_f(c[j]);
  }
  __syncthreads();

  // GEMM2: K = 128 with U2, write h_new.
#pragma unroll
  for (int nt = 0; nt < 8; ++nt) {
    float bv = c2[nt * 16 + n];
    v8f c;
#pragma unroll
    for (int j = 0; j < 8; ++j) c[j] = bv;
#pragma unroll
    for (int ks = 0; ks < 4; ++ks) {
      v16bf a = load_a_frag(smH + row * 136, ks * 32, half);
      v16bf b = *(const v16bf*)(U2f + (((long long)nt * 4 + ks) * 32 + lane) * 16);
      c = __builtin_amdgcn_wmma_f32_16x16x32_bf16(
          false, a, false, b, (short)0, c, false, false);
    }
#pragma unroll
    for (int j = 0; j < 8; ++j) {
      int r = nbase + wave * 16 + half * 8 + j;
      if (r < N) out_h[(long long)r * HID + nt * 16 + n] = c[j];
    }
  }
}

// ---------------------------------------------------------------------------
// Host launcher
// ---------------------------------------------------------------------------
extern "C" void kernel_launch(void* const* d_in, const int* in_sizes, int n_in,
                              void* d_out, int out_size, void* d_ws, size_t ws_size,
                              hipStream_t stream) {
  const float* h  = (const float*)d_in[0];
  const float* e  = (const float*)d_in[1];
  const int*   src = (const int*)d_in[2];
  const int*   dst = (const int*)d_in[3];
  const float* W1 = (const float*)d_in[4];
  const float* b1 = (const float*)d_in[5];
  const float* W2 = (const float*)d_in[6];
  const float* b2 = (const float*)d_in[7];
  const float* Wg = (const float*)d_in[8];
  const float* bg = (const float*)d_in[9];
  const float* U1 = (const float*)d_in[10];
  const float* c1 = (const float*)d_in[11];
  const float* U2 = (const float*)d_in[12];
  const float* c2 = (const float*)d_in[13];

  const int N = in_sizes[0] / HID;
  const int E = in_sizes[2];

  float* out_h = (float*)d_out;
  float* out_e = out_h + (long long)N * HID;

  // Workspace layout (all offsets 32B-aligned).
  char* ws = (char*)d_ws;
  float* agg = (float*)ws;                       // N*128 f32
  float* deg = agg + (long long)N * HID;         // N f32
  size_t off = (((size_t)N * HID + N) * 4 + 31) & ~(size_t)31;
  bf16_t* W1f = (bf16_t*)(ws + off); off += (size_t)8 * 12 * 32 * 16 * 2;
  bf16_t* W2f = (bf16_t*)(ws + off); off += (size_t)8 * 4 * 32 * 16 * 2;
  bf16_t* U1f = (bf16_t*)(ws + off); off += (size_t)8 * 8 * 32 * 16 * 2;
  bf16_t* U2f = (bf16_t*)(ws + off); off += (size_t)8 * 4 * 32 * 16 * 2;
  (void)ws_size; (void)n_in; (void)out_size;

  // 1) zero agg + deg
  long long nz = (long long)N * HID + N;
  zero_kernel<<<(int)((nz + 255) / 256), 256, 0, stream>>>(agg, nz);

  // 2) repack weights to bf16 B-fragment order
  wconv_kernel<<<(8 * 12 * 32 + 127) / 128, 128, 0, stream>>>(W1, W1f, 12);
  wconv_kernel<<<(8 * 4 * 32 + 127) / 128, 128, 0, stream>>>(W2, W2f, 4);
  wconv_kernel<<<(8 * 8 * 32 + 127) / 128, 128, 0, stream>>>(U1, U1f, 8);
  wconv_kernel<<<(8 * 4 * 32 + 127) / 128, 128, 0, stream>>>(U2, U2f, 4);

  // 3) edge messages + gating + scatter
  edge_kernel<<<(E + 63) / 64, 128, 0, stream>>>(h, e, src, dst, W1f, b1, W2f, b2,
                                                 Wg, bg, out_e, agg, deg, E);

  // 4) node update
  node_kernel<<<(N + 63) / 64, 128, 0, stream>>>(h, agg, deg, U1f, c1, U2f, c2,
                                                 out_h, N);
}
